// VoxelResBackBone8x_19628000542941
// MI455X (gfx1250) — matrix operand order repaired and checked by reading, not compiled, not run on hardware
//
#include <hip/hip_runtime.h>
#include <hip/hip_bf16.h>

#define BN_EPS 0.001f

typedef __attribute__((ext_vector_type(16))) __bf16 v16bf;
typedef __attribute__((ext_vector_type(8)))  float  v8f;

__device__ __forceinline__ unsigned bf16bits(float f) {
  unsigned u = __float_as_uint(f);
  return (u + 0x7FFFu + ((u >> 16) & 1u)) >> 16;   // round-to-nearest-even
}
__device__ __forceinline__ unsigned packbf2(float a, float b) {
  return bf16bits(a) | (bf16bits(b) << 16);
}

union ABfrag { v16bf v; unsigned u[8]; };

// ---------------------------------------------------------------------------
// Implicit-GEMM 3D conv, channels-last activations, bf16 WMMA.
// 128-thread block = 4 waves, each wave: 16 output positions (M) x NT*16
// output channels (N) -> one A-gather feeds NT WMMAs.
// Fast path (Cin >= 32): tap-outer loop, no integer division in the K loop.
// Generic path (Cin < 32): per-pair tap decode with compile-time KD/KH/KW.
// Epilogue: +bias, BN scale/shift, +residual, ReLU, *mask, optional NCDHW.
// ---------------------------------------------------------------------------
template <int NT, int KD, int KH, int KW>
__global__ __launch_bounds__(128)
void conv_wmma(const float* __restrict__ in, int Din, int Hin, int Win, int Cin, int lc,
               const unsigned short* __restrict__ wp, int Kpad, int Ktot,
               float* __restrict__ out, int Do, int Ho, int Wo, int Cout,
               const float* __restrict__ bias,
               const float* __restrict__ bng, const float* __restrict__ bnb,
               const float* __restrict__ bnm, const float* __restrict__ bnv,
               const float* __restrict__ mask,
               const float* __restrict__ resid,
               int sd, int sh, int sw, int pd, int ph, int pw, int transpose_out)
{
  const int lane = threadIdx.x & 31;
  const int wave = threadIdx.x >> 5;
  const int half = lane >> 4;          // K-half for A, row-half for B/C
  const int m    = lane & 15;          // A row / B,C column
  const int npos = Do * Ho * Wo;
  const int mt   = blockIdx.x * 4 + wave;
  if (mt * 16 >= npos) return;         // wave-uniform tail guard
  const int nb   = blockIdx.y * (NT * 16);

  // decode output coordinates for this lane's A row
  const int p  = mt * 16 + m;
  const int hw = Ho * Wo;
  int oz = p / hw; int r0 = p - oz * hw; int oy = r0 / Wo; int ox = r0 - oy * Wo;
  const int bz = oz * sd - pd, by = oy * sh - ph, bx = ox * sw - pw;
  const int cofs = half * 8;           // lane's k-offset within a 32-chunk

  const unsigned* __restrict__ brow[NT];
#pragma unroll
  for (int s = 0; s < NT; ++s)
    brow[s] = (const unsigned*)(wp + (size_t)(nb + s * 16 + m) * Kpad);

  // prefetch epilogue operands (global_prefetch_b8)
  __builtin_prefetch(mask + mt * 16, 0, 1);
  if (resid) __builtin_prefetch(resid + (size_t)(mt * 16) * Cout + nb, 0, 1);

  v8f acc[NT];
#pragma unroll
  for (int s = 0; s < NT; ++s) { v8f z = {}; acc[s] = z; }

  if (lc >= 5) {
    // ---- fast path: Cin multiple of 32, chunk boundaries align with taps ----
    int tap = 0;
    for (int kdi = 0; kdi < KD; ++kdi) {
      const int iz = bz + kdi; const bool vz = (unsigned)iz < (unsigned)Din;
      for (int khi = 0; khi < KH; ++khi) {
        const int iy = by + khi; const bool vy = (unsigned)iy < (unsigned)Hin;
        for (int kwi = 0; kwi < KW; ++kwi, ++tap) {
          const int ix = bx + kwi;
          const bool v = vz && vy && ((unsigned)ix < (unsigned)Win);
          const float* __restrict__ base =
              in + (((size_t)iz * Hin + iy) * Win + ix) * (size_t)Cin;
          for (int c0 = 0; c0 < Cin; c0 += 32) {
            ABfrag A;
#pragma unroll
            for (int j = 0; j < 8; ++j) {
              const int po = ((j & 3) * 2) + ((j >> 2) * 16) + cofs;  // k-in-chunk
              unsigned pa = 0u;
              if (v) {
                const float2 f2 = *(const float2*)(base + c0 + po);
                pa = packbf2(f2.x, f2.y);
              }
              A.u[j] = pa;
            }
            const int kb = tap * Cin + c0;
#pragma unroll
            for (int s = 0; s < NT; ++s) {
              ABfrag B;
#pragma unroll
              for (int j = 0; j < 8; ++j)
                B.u[j] = brow[s][(kb + ((j & 3) * 2) + ((j >> 2) * 16) + half * 8) >> 1];
              acc[s] = __builtin_amdgcn_wmma_f32_16x16x32_bf16(
                  false, A.v, false, B.v, (short)0, acc[s], false, false);
            }
          }
        }
      }
    }
  } else {
    // ---- generic path: per-pair decode, constant divisors via template ----
    const int cmask = (1 << lc) - 1;
    for (int k0 = 0; k0 < Kpad; k0 += 32) {
      ABfrag A;
#pragma unroll
      for (int j = 0; j < 8; ++j) {
        const int kk = k0 + ((j & 3) * 2) + ((j >> 2) * 16) + cofs;
        unsigned pa = 0u;
        if (kk < Ktot) {
          const int cin = kk & cmask;
          const int tap = kk >> lc;
          const int kdi = tap / (KH * KW); const int r2 = tap - kdi * (KH * KW);
          const int khi = r2 / KW;         const int kwi = r2 - khi * KW;
          const int iz = bz + kdi, iy = by + khi, ix = bx + kwi;
          if ((unsigned)iz < (unsigned)Din && (unsigned)iy < (unsigned)Hin &&
              (unsigned)ix < (unsigned)Win) {
            const float2 f2 = *(const float2*)(in +
                (((size_t)iz * Hin + iy) * Win + ix) * (size_t)Cin + cin);
            pa = packbf2(f2.x, f2.y);
          }
        }
        A.u[j] = pa;
      }
#pragma unroll
      for (int s = 0; s < NT; ++s) {
        ABfrag B;
#pragma unroll
        for (int j = 0; j < 8; ++j)
          B.u[j] = brow[s][(k0 + ((j & 3) * 2) + ((j >> 2) * 16) + half * 8) >> 1];
        acc[s] = __builtin_amdgcn_wmma_f32_16x16x32_bf16(
            false, A.v, false, B.v, (short)0, acc[s], false, false);
      }
    }
  }

  // fused epilogue
#pragma unroll
  for (int s = 0; s < NT; ++s) {
    const int co = nb + s * 16 + m;
    const float scale = bng[co] * rsqrtf(bnv[co] + BN_EPS);
    const float shift = bnb[co] - bnm[co] * scale;
    const float bs = bias ? bias[co] : 0.0f;
#pragma unroll
    for (int r = 0; r < 8; ++r) {
      const int pr = mt * 16 + r + half * 8;    // C/D layout: VGPR r -> M = r + 8*half
      float y = (acc[s][r] + bs) * scale + shift;
      if (resid) y += resid[(size_t)pr * Cout + co];
      y = fmaxf(y, 0.0f);
      y *= mask[pr];
      if (transpose_out) out[(size_t)co * npos + pr] = y;    // NCDHW final output
      else               out[(size_t)pr * Cout + co] = y;    // channels-last
    }
  }
}

// ---------------------------------------------------------------------------
// Weight repack: OIDHW f32 -> [Cout][Kpad] bf16 with k = tap*Cin + cin.
// ---------------------------------------------------------------------------
__global__ void pack_w(const float* __restrict__ w, unsigned short* __restrict__ dst,
                       int Cout, int Cin, int lc, int KV, int Kpad, int Ktot)
{
  int t = blockIdx.x * 256 + threadIdx.x;
  if (t >= Cout * Kpad) return;
  int co = t / Kpad; int k = t - co * Kpad;
  float v = 0.0f;
  if (k < Ktot) {
    int cin = k & (Cin - 1);
    int tap = k >> lc;
    v = w[((size_t)co * Cin + cin) * KV + tap];
  }
  dst[t] = (unsigned short)bf16bits(v);
}

__global__ void zero_f32(float* __restrict__ p, size_t n) {
  size_t i = (size_t)blockIdx.x * 256 + threadIdx.x;
  size_t st = (size_t)gridDim.x * 256;
  for (; i < n; i += st) p[i] = 0.0f;
}

__global__ void scatter_k(const float* __restrict__ vf, const int* __restrict__ vc,
                          float* __restrict__ dense4, float* __restrict__ mask1,
                          int nvox, int H, int Wd)
{
  int t = blockIdx.x * 256 + threadIdx.x;
  if (t >= nvox) return;
  int z = vc[t * 4 + 1], y = vc[t * 4 + 2], x = vc[t * 4 + 3];
  size_t p = ((size_t)z * H + y) * Wd + x;
  dense4[p * 4 + 0] = vf[t * 4 + 0];
  dense4[p * 4 + 1] = vf[t * 4 + 1];
  dense4[p * 4 + 2] = vf[t * 4 + 2];
  dense4[p * 4 + 3] = vf[t * 4 + 3];
  mask1[p] = 1.0f;
}

__global__ void dilate_k(const float* __restrict__ mi, int Di, int Hi, int Wi,
                         float* __restrict__ mo, int Do, int Ho, int Wo,
                         int kd, int kh, int kw, int sd, int sh, int sw,
                         int pd, int ph, int pw)
{
  int p = blockIdx.x * 256 + threadIdx.x;
  int npos = Do * Ho * Wo;
  if (p >= npos) return;
  int oz = p / (Ho * Wo); int r = p - oz * Ho * Wo; int oy = r / Wo; int ox = r - oy * Wo;
  float v = 0.0f;
  for (int a = 0; a < kd; ++a) {
    int iz = oz * sd - pd + a; if ((unsigned)iz >= (unsigned)Di) continue;
    for (int b = 0; b < kh; ++b) {
      int iy = oy * sh - ph + b; if ((unsigned)iy >= (unsigned)Hi) continue;
      for (int c = 0; c < kw; ++c) {
        int ix = ox * sw - pw + c; if ((unsigned)ix >= (unsigned)Wi) continue;
        if (mi[((size_t)iz * Hi + iy) * Wi + ix] > 0.0f) v = 1.0f;
      }
    }
  }
  mo[p] = v;
}

// ---------------------------------------------------------------------------
extern "C" void kernel_launch(void* const* d_in, const int* in_sizes, int n_in,
                              void* d_out, int out_size, void* d_ws, size_t ws_size,
                              hipStream_t stream) {
  (void)in_sizes; (void)n_in; (void)out_size; (void)ws_size;
  float* W = (float*)d_ws;
  auto F = [&](int i) { return (const float*)d_in[i]; };

  // workspace layout (float offsets)
  const size_t o_dense4 = 0;                     // 41*160*160*4
  const size_t o_mask1  = 4198400;               // 41*160*160
  const size_t o_mask2  = o_mask1 + 1049600;     // 21*80*80
  const size_t o_mask3  = o_mask2 + 134400;      // 11*40*40
  const size_t o_mask4  = o_mask3 + 17600;       // 5*20*20
  const size_t o_mask5  = o_mask4 + 2000;        // 2*20*20
  const size_t o_reg0   = o_mask5 + 800;         // 41*160*160*16
  const size_t o_reg1   = o_reg0 + 16793600;
  const size_t o_wp     = o_reg1 + 16793600;
  unsigned short* wpack = (unsigned short*)(W + o_wp);
  size_t wpo = 0;

  // scatter voxels -> dense NDHWC(4) grid + mask
  zero_f32<<<2048, 256, 0, stream>>>(W + o_dense4, (size_t)5248000);
  scatter_k<<<(50000 + 255) / 256, 256, 0, stream>>>(
      F(0), (const int*)d_in[1], W + o_dense4, W + o_mask1, 50000, 160, 160);

  // mask pyramid
  dilate_k<<<(134400 + 255) / 256, 256, 0, stream>>>(W + o_mask1, 41, 160, 160,
      W + o_mask2, 21, 80, 80, 3, 3, 3, 2, 2, 2, 1, 1, 1);
  dilate_k<<<(17600 + 255) / 256, 256, 0, stream>>>(W + o_mask2, 21, 80, 80,
      W + o_mask3, 11, 40, 40, 3, 3, 3, 2, 2, 2, 1, 1, 1);
  dilate_k<<<(2000 + 255) / 256, 256, 0, stream>>>(W + o_mask3, 11, 40, 40,
      W + o_mask4, 5, 20, 20, 3, 3, 3, 2, 2, 2, 0, 1, 1);
  dilate_k<<<(800 + 255) / 256, 256, 0, stream>>>(W + o_mask4, 5, 20, 20,
      W + o_mask5, 2, 20, 20, 3, 1, 1, 2, 1, 1, 0, 0, 0);

  auto pack = [&](int widx, int Cout, int Cin, int lc, int KV, int Kpad,
                  int Ktot) -> const unsigned short* {
    unsigned short* dst = wpack + wpo;
    wpo += (size_t)Cout * Kpad;
    int tot = Cout * Kpad;
    pack_w<<<(tot + 255) / 256, 256, 0, stream>>>(F(widx), dst, Cout, Cin, lc, KV, Kpad, Ktot);
    return dst;
  };

  auto conv = [&](const float* in, int Di, int Hi, int Wi, int Cin, int lc,
                  const unsigned short* wp, int Kpad, int Ktot,
                  float* out, int Do, int Ho, int Wo, int Cout,
                  const float* bias, int bnIdx, const float* mask, const float* resid,
                  int k311, int sd, int sh, int sw, int pd, int ph, int pw, int tr) {
    int nt = (Cout >= 64) ? 4 : (Cout == 32 ? 2 : 1);
    int mtiles = Do * Ho * Wo / 16;
    dim3 g((unsigned)((mtiles + 3) / 4), (unsigned)(Cout / (nt * 16)));
    if (k311)
      conv_wmma<4, 3, 1, 1><<<g, 128, 0, stream>>>(in, Di, Hi, Wi, Cin, lc, wp, Kpad, Ktot,
          out, Do, Ho, Wo, Cout, bias, F(bnIdx), F(bnIdx + 1), F(bnIdx + 2), F(bnIdx + 3),
          mask, resid, sd, sh, sw, pd, ph, pw, tr);
    else if (nt == 1)
      conv_wmma<1, 3, 3, 3><<<g, 128, 0, stream>>>(in, Di, Hi, Wi, Cin, lc, wp, Kpad, Ktot,
          out, Do, Ho, Wo, Cout, bias, F(bnIdx), F(bnIdx + 1), F(bnIdx + 2), F(bnIdx + 3),
          mask, resid, sd, sh, sw, pd, ph, pw, tr);
    else if (nt == 2)
      conv_wmma<2, 3, 3, 3><<<g, 128, 0, stream>>>(in, Di, Hi, Wi, Cin, lc, wp, Kpad, Ktot,
          out, Do, Ho, Wo, Cout, bias, F(bnIdx), F(bnIdx + 1), F(bnIdx + 2), F(bnIdx + 3),
          mask, resid, sd, sh, sw, pd, ph, pw, tr);
    else
      conv_wmma<4, 3, 3, 3><<<g, 128, 0, stream>>>(in, Di, Hi, Wi, Cin, lc, wp, Kpad, Ktot,
          out, Do, Ho, Wo, Cout, bias, F(bnIdx), F(bnIdx + 1), F(bnIdx + 2), F(bnIdx + 3),
          mask, resid, sd, sh, sw, pd, ph, pw, tr);
  };

  auto resblock = [&](int pi, int Di, int Hi, int Wi, int C, int lc,
                      float* P, float* Q, const float* mask) {
    int KV = 27, Ktot = 27 * C, Kpad = (Ktot + 31) & ~31;
    const unsigned short* w1 = pack(pi, C, C, lc, KV, Kpad, Ktot);
    const unsigned short* w2 = pack(pi + 6, C, C, lc, KV, Kpad, Ktot);
    conv(P, Di, Hi, Wi, C, lc, w1, Kpad, Ktot, Q, Di, Hi, Wi, C,
         F(pi + 1), pi + 2, mask, nullptr, 0, 1, 1, 1, 1, 1, 1, 0);
    conv(Q, Di, Hi, Wi, C, lc, w2, Kpad, Ktot, P, Di, Hi, Wi, C,
         F(pi + 7), pi + 8, mask, P, 0, 1, 1, 1, 1, 1, 1, 0);   // in-place residual
  };

  float* P1 = W + o_reg0; float* Q1 = W + o_reg1;

  // conv_input: 4 -> 16, idx: w=2, bn=3..6 (no bias)
  { int Ktot = 108, Kpad = 128;
    const unsigned short* wci = pack(2, 16, 4, 2, 27, Kpad, Ktot);
    conv(W + o_dense4, 41, 160, 160, 4, 2, wci, Kpad, Ktot, P1, 41, 160, 160, 16,
         nullptr, 3, W + o_mask1, nullptr, 0, 1, 1, 1, 1, 1, 1, 0); }
  resblock(7,  41, 160, 160, 16, 4, P1, Q1, W + o_mask1);
  resblock(19, 41, 160, 160, 16, 4, P1, Q1, W + o_mask1);

  // down2: 16 -> 32, stride 2, pad 1. idx: w=31, bn=32..35
  float* P2 = W + o_reg1; float* Q2 = W + o_reg1 + 4300800;
  { int Ktot = 432, Kpad = 448;
    const unsigned short* wd = pack(31, 32, 16, 4, 27, Kpad, Ktot);
    conv(P1, 41, 160, 160, 16, 4, wd, Kpad, Ktot, P2, 21, 80, 80, 32,
         nullptr, 32, W + o_mask2, nullptr, 0, 2, 2, 2, 1, 1, 1, 0); }
  resblock(36, 21, 80, 80, 32, 5, P2, Q2, W + o_mask2);
  resblock(48, 21, 80, 80, 32, 5, P2, Q2, W + o_mask2);

  // down3: 32 -> 64. idx: w=60, bn=61..64
  float* P3 = W + o_reg0; float* Q3 = W + o_reg0 + 1126400;
  { int Ktot = 864, Kpad = 864;
    const unsigned short* wd = pack(60, 64, 32, 5, 27, Kpad, Ktot);
    conv(P2, 21, 80, 80, 32, 5, wd, Kpad, Ktot, P3, 11, 40, 40, 64,
         nullptr, 61, W + o_mask3, nullptr, 0, 2, 2, 2, 1, 1, 1, 0); }
  resblock(65, 11, 40, 40, 64, 6, P3, Q3, W + o_mask3);
  resblock(77, 11, 40, 40, 64, 6, P3, Q3, W + o_mask3);

  // down4: 64 -> 128, pad (0,1,1). idx: w=89, bn=90..93
  float* P4 = W + o_reg1; float* Q4 = W + o_reg1 + 256000;
  { int Ktot = 1728, Kpad = 1728;
    const unsigned short* wd = pack(89, 128, 64, 6, 27, Kpad, Ktot);
    conv(P3, 11, 40, 40, 64, 6, wd, Kpad, Ktot, P4, 5, 20, 20, 128,
         nullptr, 90, W + o_mask4, nullptr, 0, 2, 2, 2, 0, 1, 1, 0); }
  resblock(94,  5, 20, 20, 128, 7, P4, Q4, W + o_mask4);
  resblock(106, 5, 20, 20, 128, 7, P4, Q4, W + o_mask4);

  // conv_out: 128 -> 128, k(3,1,1), stride (2,1,1), pad 0, NCDHW store to d_out.
  // idx: w=118, bn=119..122
  { int Ktot = 384, Kpad = 384;
    const unsigned short* wd = pack(118, 128, 128, 7, 3, Kpad, Ktot);
    conv(P4, 5, 20, 20, 128, 7, wd, Kpad, Ktot, (float*)d_out, 2, 20, 20, 128,
         nullptr, 119, W + o_mask5, nullptr, 1, 2, 1, 1, 0, 0, 0, 1); }
}
